// Seq2SeqTransformer_82437602279587
// MI455X (gfx1250) — matrix operand order repaired and checked by reading, not compiled, not run on hardware
//
#include <hip/hip_runtime.h>

#define BB 8
#define SS 512
#define DD 512
#define HH 8
#define LL 6
#define FF 2048
#define VV 32000
#define DKK 64
#define NT (BB*SS) /* 4096 tokens per stream */

typedef __attribute__((ext_vector_type(16))) __bf16 v16bf;
typedef __attribute__((ext_vector_type(8)))  float  v8f;

union Frag { uint4 u[2]; v16bf v; };

__device__ __forceinline__ unsigned short f2bf(float f) {
    unsigned u = __float_as_uint(f);
    unsigned r = u + 0x7FFFu + ((u >> 16) & 1u);   // round-to-nearest-even
    return (unsigned short)(r >> 16);
}
__device__ __forceinline__ unsigned pk2(float a, float b) {
    return (unsigned)f2bf(a) | ((unsigned)f2bf(b) << 16);
}
__device__ __forceinline__ v8f wmma_bf16(const Frag& a, const Frag& b, v8f c) {
    return __builtin_amdgcn_wmma_f32_16x16x32_bf16(false, a.v, false, b.v,
                                                   (short)0, c, false, false);
}

// ---------------------------------------------------------------------------
// Embedding * sqrt(D) + positional encoding (computed on the fly)
// ---------------------------------------------------------------------------
__global__ __launch_bounds__(256) void embed_pe_kernel(
    const int* __restrict__ tok, const float* __restrict__ emb,
    float* __restrict__ out)
{
    const int t  = blockIdx.x;          // token index 0..NT-1
    const int s  = t & (SS - 1);        // position within sequence
    const int tk = tok[t];
    for (int i = threadIdx.x; i < DD; i += 256) {
        float ang = (float)s * __powf(10000.0f, -2.0f * (float)i / (float)DD);
        float pe  = (i & 1) ? __cosf(ang) : __sinf(ang);
        out[(size_t)t * DD + i] = emb[(size_t)tk * DD + i] * 22.62741699796952f + pe;
    }
}

// ---------------------------------------------------------------------------
// LayerNorm over last dim (D=512), one block per token row
// ---------------------------------------------------------------------------
__global__ __launch_bounds__(256) void layernorm_kernel(
    const float* __restrict__ x, const float* __restrict__ g,
    const float* __restrict__ b, float* __restrict__ o)
{
    __shared__ float red[256];
    const int t = blockIdx.x;
    const float* xr = x + (size_t)t * DD;
    const float v0 = xr[threadIdx.x];
    const float v1 = xr[threadIdx.x + 256];

    red[threadIdx.x] = v0 + v1;
    __syncthreads();
    for (int s = 128; s > 0; s >>= 1) {
        if (threadIdx.x < s) red[threadIdx.x] += red[threadIdx.x + s];
        __syncthreads();
    }
    const float mu = red[0] * (1.0f / DD);
    __syncthreads();

    const float d0 = v0 - mu, d1 = v1 - mu;
    red[threadIdx.x] = d0 * d0 + d1 * d1;
    __syncthreads();
    for (int s = 128; s > 0; s >>= 1) {
        if (threadIdx.x < s) red[threadIdx.x] += red[threadIdx.x + s];
        __syncthreads();
    }
    const float var  = red[0] * (1.0f / DD);
    const float rstd = rsqrtf(var + 1e-6f);

    float* orow = o + (size_t)t * DD;
    orow[threadIdx.x]       = d0 * rstd * g[threadIdx.x]       + b[threadIdx.x];
    orow[threadIdx.x + 256] = d1 * rstd * g[threadIdx.x + 256] + b[threadIdx.x + 256];
}

// ---------------------------------------------------------------------------
// C[M,N] = (relu?)(A[M,K] @ W[K,N] + bias[N]) (+ resid[M,N])
// 128x128 tile per block, 256 threads = 8 waves, BK=32, bf16 WMMA, f32 acc.
// Double-buffered LDS tiles: global fetch of tile k+1 overlaps WMMA of tile k.
// Grid: (N/128, M/128). M,N multiples of 128; K multiple of 32.
// ---------------------------------------------------------------------------
__global__ __launch_bounds__(256) void gemm_wmma_kernel(
    const float* __restrict__ A, int lda,
    const float* __restrict__ W, int ldw,
    const float* __restrict__ bias,
    const float* __restrict__ resid,
    float* __restrict__ C, int ldc,
    int K, int relu)
{
    __shared__ __align__(16) unsigned short As[2][128 * 40]; // 32 data + 8 pad / row
    __shared__ __align__(16) unsigned short Bs[2][128 * 40]; // transposed: row=n

    const int tid  = threadIdx.x;
    const int wave = tid >> 5;
    const int lane = tid & 31;
    const int m16  = lane & 15;
    const int kh   = lane >> 4;
    const int brow = blockIdx.y * 128;
    const int bcol = blockIdx.x * 128;

    v8f acc[8];
#pragma unroll
    for (int j = 0; j < 8; ++j) acc[j] = {};

    float4 a_st[4];   // staged A tile (this thread's share)
    float4 w_st[4];   // staged W tile

    auto load_tiles = [&](int kb) {
#pragma unroll
        for (int i = 0; i < 4; ++i) {
            const int li = tid + i * 256;
            a_st[i] = *reinterpret_cast<const float4*>(
                A + (size_t)(brow + (li >> 3)) * lda + kb + ((li & 7) << 2));
        }
#pragma unroll
        for (int i = 0; i < 4; ++i) {
            const int li = tid + i * 256;
            w_st[i] = *reinterpret_cast<const float4*>(
                W + (size_t)(kb + (li >> 5)) * ldw + bcol + ((li & 31) << 2));
        }
        // prefetch the tile after next into cache (speculative, b8 granule)
        if (kb + 64 < K) {
            __builtin_prefetch(W + (size_t)(kb + 32 + (tid >> 5)) * ldw + bcol
                                 + ((tid & 31) << 2), 0, 1);
            __builtin_prefetch(A + (size_t)(brow + (tid >> 3)) * lda + kb + 64
                                 + ((tid & 7) << 2), 0, 1);
        }
    };
    auto store_tiles = [&](int buf) {
        unsigned short* as = As[buf];
        unsigned short* bs = Bs[buf];
#pragma unroll
        for (int i = 0; i < 4; ++i) {
            const int li = tid + i * 256;
            const int r  = li >> 3;
            const int c4 = (li & 7) << 2;
            uint2 p; p.x = pk2(a_st[i].x, a_st[i].y); p.y = pk2(a_st[i].z, a_st[i].w);
            *reinterpret_cast<uint2*>(&as[r * 40 + c4]) = p;
        }
#pragma unroll
        for (int i = 0; i < 4; ++i) {
            const int li = tid + i * 256;
            const int k  = li >> 5;
            const int n4 = (li & 31) << 2;
            bs[(n4 + 0) * 40 + k] = f2bf(w_st[i].x);
            bs[(n4 + 1) * 40 + k] = f2bf(w_st[i].y);
            bs[(n4 + 2) * 40 + k] = f2bf(w_st[i].z);
            bs[(n4 + 3) * 40 + k] = f2bf(w_st[i].w);
        }
    };

    // prologue: tile 0 -> buffer 0
    load_tiles(0);
    store_tiles(0);
    __syncthreads();

    const int nk = K >> 5;
    for (int it = 0; it < nk; ++it) {
        const int cur = it & 1;
        const bool more = (it + 1) < nk;
        if (more) load_tiles((it + 1) << 5);   // overlaps with compute below

        const unsigned short* as = As[cur];
        const unsigned short* bs = Bs[cur];

        // A fragment: lane(m16,kh): k runs {kh*8..+7} and {16+kh*8..+7}
        Frag fa;
        const unsigned short* ap = &as[(wave * 16 + m16) * 40 + kh * 8];
        fa.u[0] = *reinterpret_cast<const uint4*>(ap);
        fa.u[1] = *reinterpret_cast<const uint4*>(ap + 16);

        // B fragments in batches of 4 so LDS loads pipeline ahead of the WMMAs
#pragma unroll
        for (int jj = 0; jj < 2; ++jj) {
            Frag fb[4];
#pragma unroll
            for (int j = 0; j < 4; ++j) {
                const unsigned short* bp =
                    &bs[((jj * 4 + j) * 16 + m16) * 40 + kh * 8];
                fb[j].u[0] = *reinterpret_cast<const uint4*>(bp);
                fb[j].u[1] = *reinterpret_cast<const uint4*>(bp + 16);
            }
#pragma unroll
            for (int j = 0; j < 4; ++j)
                acc[jj * 4 + j] = wmma_bf16(fa, fb[j], acc[jj * 4 + j]);
        }

        if (more) store_tiles(1 - cur);
        __syncthreads();
    }

    // ---- epilogue: bias (+relu) (+residual) ----
    const int grow0 = brow + wave * 16 + kh * 8;
    float bv[8];
#pragma unroll
    for (int j = 0; j < 8; ++j) bv[j] = bias[bcol + j * 16 + m16];

#pragma unroll
    for (int v = 0; v < 8; ++v) {
        const size_t base = (size_t)(grow0 + v) * ldc + bcol + m16;
#pragma unroll
        for (int j = 0; j < 8; ++j) {
            float val = acc[j][v] + bv[j];
            if (relu) val = fmaxf(val, 0.0f);
            if (resid) val += resid[base + j * 16];
            C[base + j * 16] = val;
        }
    }
}

// ---------------------------------------------------------------------------
// Fused attention for one (b, h, 16-query block):
//   scores = QK^T/sqrt(dk), mask, softmax, out = P V
// Q/K/V/O layouts: [B*S, D] with head h occupying cols h*64..h*64+63.
// Grid: (S/16, B*H), 128 threads = 4 waves.
// ---------------------------------------------------------------------------
__global__ __launch_bounds__(128) void attention_kernel(
    const float* __restrict__ Qb, const float* __restrict__ Kb,
    const float* __restrict__ Vb, float* __restrict__ Ob,
    const int* __restrict__ mtok, int causal)
{
    __shared__ __align__(16) unsigned short qs[16 * 72];   // 16 q rows x 64 dk (+pad)
    __shared__ __align__(16) unsigned short kt[128 * 72];  // 128 keys x 64 dk (+pad)
    __shared__ __align__(16) unsigned short vt[64 * 40];   // 64 dk x 32 keys (+pad)
    __shared__ __align__(16) float sc[16 * 520];           // scores 16 x 512 (+pad)
    __shared__ float redA[16 * 8];
    __shared__ float redB[16 * 8];
    __shared__ int   kval[128];

    const int tid  = threadIdx.x;
    const int wave = tid >> 5;
    const int lane = tid & 31;
    const int m16  = lane & 15;
    const int kh   = lane >> 4;
    const int q0   = blockIdx.x * 16;
    const int b    = blockIdx.y >> 3;   // H = 8
    const int h    = blockIdx.y & 7;
    const size_t tokbase = (size_t)b * SS;

    // ---- load Q tile 16x64 -> bf16 LDS ----
#pragma unroll
    for (int i = 0; i < 2; ++i) {
        int li = tid + i * 128;            // 256 float4 total
        int r  = li >> 4;
        int c4 = (li & 15) << 2;
        const float4 f = *reinterpret_cast<const float4*>(
            Qb + (tokbase + q0 + r) * DD + h * DKK + c4);
        uint2 p; p.x = pk2(f.x, f.y); p.y = pk2(f.z, f.w);
        *reinterpret_cast<uint2*>(&qs[r * 72 + c4]) = p;
    }

    // ---- scores: loop over 4 key blocks of 128 ----
    for (int kb = 0; kb < 4; ++kb) {
#pragma unroll
        for (int i = 0; i < 16; ++i) {
            int li = tid + i * 128;        // 2048 float4 total
            int r  = li >> 4;
            int c4 = (li & 15) << 2;
            const float4 f = *reinterpret_cast<const float4*>(
                Kb + (tokbase + kb * 128 + r) * DD + h * DKK + c4);
            uint2 p; p.x = pk2(f.x, f.y); p.y = pk2(f.z, f.w);
            *reinterpret_cast<uint2*>(&kt[r * 72 + c4]) = p;
        }
        kval[tid] = (mtok[tokbase + kb * 128 + tid] != 0);
        __syncthreads();

        // A fragments for both 32-wide k-steps (dk 0..31 and 32..63)
        Frag fa0, fa1;
        {
            const unsigned short* ap = &qs[m16 * 72 + kh * 8];
            fa0.u[0] = *reinterpret_cast<const uint4*>(ap);
            fa0.u[1] = *reinterpret_cast<const uint4*>(ap + 16);
            fa1.u[0] = *reinterpret_cast<const uint4*>(ap + 32);
            fa1.u[1] = *reinterpret_cast<const uint4*>(ap + 48);
        }
#pragma unroll
        for (int si = 0; si < 2; ++si) {
            const int st = wave * 2 + si;               // 8 subtiles / 4 waves
            const unsigned short* bp = &kt[(st * 16 + m16) * 72 + kh * 8];
            Frag fb0, fb1;
            fb0.u[0] = *reinterpret_cast<const uint4*>(bp);
            fb0.u[1] = *reinterpret_cast<const uint4*>(bp + 16);
            fb1.u[0] = *reinterpret_cast<const uint4*>(bp + 32);
            fb1.u[1] = *reinterpret_cast<const uint4*>(bp + 48);
            v8f c = {};
            c = wmma_bf16(fa0, fb0, c);
            c = wmma_bf16(fa1, fb1, c);

            const int klocal = st * 16 + m16;
            const int key    = kb * 128 + klocal;
            const int pad_ok = kval[klocal];
#pragma unroll
            for (int v = 0; v < 8; ++v) {
                const int qrow = q0 + kh * 8 + v;
                const bool ok  = pad_ok && (!causal || key <= qrow);
                sc[(kh * 8 + v) * 520 + key] = ok ? c[v] * 0.125f : -1e9f;
            }
        }
        __syncthreads();
    }

    // ---- softmax over each of the 16 rows (512 cols), 8 threads/row ----
    {
        const int row = tid >> 3;
        const int sub = tid & 7;
        float* srow = &sc[row * 520];
        float mx = -3.0e38f;
        for (int i = 0; i < 64; ++i) mx = fmaxf(mx, srow[sub * 64 + i]);
        redA[row * 8 + sub] = mx;
        __syncthreads();
        mx = redA[row * 8];
#pragma unroll
        for (int i = 1; i < 8; ++i) mx = fmaxf(mx, redA[row * 8 + i]);
        float sum = 0.0f;
        for (int i = 0; i < 64; ++i) {
            float e = __expf(srow[sub * 64 + i] - mx);
            srow[sub * 64 + i] = e;
            sum += e;
        }
        redB[row * 8 + sub] = sum;
        __syncthreads();
        sum = 0.0f;
#pragma unroll
        for (int i = 0; i < 8; ++i) sum += redB[row * 8 + i];
        const float inv = 1.0f / sum;
        for (int i = 0; i < 64; ++i) srow[sub * 64 + i] *= inv;
    }
    __syncthreads();

    // ---- out = P @ V : 16x64 tile, wave w owns dk subtile w ----
    v8f co = {};
    for (int ks = 0; ks < 16; ++ks) {
        // load V block 32 keys x 64 dk, transpose into vt[dk][key]
#pragma unroll
        for (int i = 0; i < 4; ++i) {
            int li = tid + i * 128;        // 512 float4 total
            int r  = li >> 4;
            int c4 = (li & 15) << 2;
            const float4 f = *reinterpret_cast<const float4*>(
                Vb + (tokbase + ks * 32 + r) * DD + h * DKK + c4);
            vt[(c4 + 0) * 40 + r] = f2bf(f.x);
            vt[(c4 + 1) * 40 + r] = f2bf(f.y);
            vt[(c4 + 2) * 40 + r] = f2bf(f.z);
            vt[(c4 + 3) * 40 + r] = f2bf(f.w);
        }
        __syncthreads();

        // A fragment from probabilities (fp32 LDS -> bf16)
        Frag fa;
        const float* prow = &sc[m16 * 520 + ks * 32 + kh * 8];
        const float4 p0 = *reinterpret_cast<const float4*>(prow);
        const float4 p1 = *reinterpret_cast<const float4*>(prow + 4);
        const float4 p2 = *reinterpret_cast<const float4*>(prow + 16);
        const float4 p3 = *reinterpret_cast<const float4*>(prow + 20);
        fa.u[0] = make_uint4(pk2(p0.x, p0.y), pk2(p0.z, p0.w),
                             pk2(p1.x, p1.y), pk2(p1.z, p1.w));
        fa.u[1] = make_uint4(pk2(p2.x, p2.y), pk2(p2.z, p2.w),
                             pk2(p3.x, p3.y), pk2(p3.z, p3.w));

        // B fragment from transposed V
        Frag fb;
        const unsigned short* bp = &vt[(wave * 16 + m16) * 40 + kh * 8];
        fb.u[0] = *reinterpret_cast<const uint4*>(bp);
        fb.u[1] = *reinterpret_cast<const uint4*>(bp + 16);

        co = wmma_bf16(fa, fb, co);
        __syncthreads();
    }

#pragma unroll
    for (int v = 0; v < 8; ++v) {
        const int tokr = q0 + kh * 8 + v;
        Ob[(tokbase + tokr) * DD + h * DKK + wave * 16 + m16] = co[v];
    }
}

// ---------------------------------------------------------------------------
// In-place log_softmax over V=32000, one block per token row
// ---------------------------------------------------------------------------
__global__ __launch_bounds__(256) void logsoftmax_kernel(float* __restrict__ logits)
{
    __shared__ float red[256];
    float* row = logits + (size_t)blockIdx.x * VV;

    float mx = -3.0e38f;
    for (int i = threadIdx.x; i < VV; i += 256) mx = fmaxf(mx, row[i]);
    red[threadIdx.x] = mx;
    __syncthreads();
    for (int s = 128; s > 0; s >>= 1) {
        if (threadIdx.x < s) red[threadIdx.x] = fmaxf(red[threadIdx.x], red[threadIdx.x + s]);
        __syncthreads();
    }
    mx = red[0];
    __syncthreads();

    float sum = 0.0f;
    for (int i = threadIdx.x; i < VV; i += 256) sum += __expf(row[i] - mx);
    red[threadIdx.x] = sum;
    __syncthreads();
    for (int s = 128; s > 0; s >>= 1) {
        if (threadIdx.x < s) red[threadIdx.x] += red[threadIdx.x + s];
        __syncthreads();
    }
    const float lse = mx + __logf(red[0]);

    for (int i = threadIdx.x; i < VV; i += 256) row[i] = row[i] - lse;
}

// ---------------------------------------------------------------------------
// Host orchestration
// ---------------------------------------------------------------------------
extern "C" void kernel_launch(void* const* d_in, const int* in_sizes, int n_in,
                              void* d_out, int out_size, void* d_ws, size_t ws_size,
                              hipStream_t stream)
{
    (void)in_sizes; (void)n_in; (void)out_size; (void)ws_size;

    const int*   src      = (const int*)  d_in[0];
    const int*   trg      = (const int*)  d_in[1];
    const float* src_emb  = (const float*)d_in[2];
    const float* trg_emb  = (const float*)d_in[3];
    const float* e_wq  = (const float*)d_in[4];
    const float* e_bq  = (const float*)d_in[5];
    const float* e_wk  = (const float*)d_in[6];
    const float* e_bk  = (const float*)d_in[7];
    const float* e_wv  = (const float*)d_in[8];
    const float* e_bv  = (const float*)d_in[9];
    const float* e_wo  = (const float*)d_in[10];
    const float* e_bo  = (const float*)d_in[11];
    const float* dsa_wq = (const float*)d_in[12];
    const float* dsa_bq = (const float*)d_in[13];
    const float* dsa_wk = (const float*)d_in[14];
    const float* dsa_bk = (const float*)d_in[15];
    const float* dsa_wv = (const float*)d_in[16];
    const float* dsa_bv = (const float*)d_in[17];
    const float* dsa_wo = (const float*)d_in[18];
    const float* dsa_bo = (const float*)d_in[19];
    const float* dca_wq = (const float*)d_in[20];
    const float* dca_bq = (const float*)d_in[21];
    const float* dca_wk = (const float*)d_in[22];
    const float* dca_bk = (const float*)d_in[23];
    const float* dca_wv = (const float*)d_in[24];
    const float* dca_bv = (const float*)d_in[25];
    const float* dca_wo = (const float*)d_in[26];
    const float* dca_bo = (const float*)d_in[27];
    const float* e_ln1_g = (const float*)d_in[28];
    const float* e_ln1_b = (const float*)d_in[29];
    const float* e_ln2_g = (const float*)d_in[30];
    const float* e_ln2_b = (const float*)d_in[31];
    const float* d_ln1_g = (const float*)d_in[32];
    const float* d_ln1_b = (const float*)d_in[33];
    const float* d_ln2_g = (const float*)d_in[34];
    const float* d_ln2_b = (const float*)d_in[35];
    const float* d_ln3_g = (const float*)d_in[36];
    const float* d_ln3_b = (const float*)d_in[37];
    const float* e_w1 = (const float*)d_in[38];
    const float* e_b1 = (const float*)d_in[39];
    const float* e_w2 = (const float*)d_in[40];
    const float* e_b2 = (const float*)d_in[41];
    const float* e_lnf_g = (const float*)d_in[42];
    const float* e_lnf_b = (const float*)d_in[43];
    const float* d_w1 = (const float*)d_in[44];
    const float* d_b1 = (const float*)d_in[45];
    const float* d_w2 = (const float*)d_in[46];
    const float* d_b2 = (const float*)d_in[47];
    const float* d_lnf_g = (const float*)d_in[48];
    const float* d_lnf_b = (const float*)d_in[49];
    const float* out_w = (const float*)d_in[50];
    const float* out_b = (const float*)d_in[51];

    float* X    = (float*)d_ws;
    float* Y    = X    + (size_t)NT * DD;
    float* T1   = Y    + (size_t)NT * DD;
    float* Q    = T1   + (size_t)NT * DD;
    float* Kbuf = Q    + (size_t)NT * DD;
    float* Vbuf = Kbuf + (size_t)NT * DD;
    float* AO   = Vbuf + (size_t)NT * DD;
    float* EOUT = AO   + (size_t)NT * DD;
    float* HID  = EOUT + (size_t)NT * DD;
    float* OUT  = (float*)d_out;

    const dim3 g512(DD / 128, NT / 128);   // (4, 32)
    const dim3 gF  (FF / 128, NT / 128);   // (16, 32)
    const dim3 gV  (VV / 128, NT / 128);   // (250, 32)
    const dim3 gAtt(SS / 16, BB * HH);     // (32, 64)

    embed_pe_kernel<<<NT, 256, 0, stream>>>(src, src_emb, X);
    embed_pe_kernel<<<NT, 256, 0, stream>>>(trg, trg_emb, Y);

    // ---------------- encoder ----------------
    for (int l = 0; l < LL; ++l) {
        const size_t wdd = (size_t)l * DD * DD;
        layernorm_kernel<<<NT, 256, 0, stream>>>(X, e_ln1_g + l * DD, e_ln1_b + l * DD, T1);
        gemm_wmma_kernel<<<g512, 256, 0, stream>>>(T1, DD, e_wq + wdd, DD, e_bq + l * DD, nullptr, Q,    DD, DD, 0);
        gemm_wmma_kernel<<<g512, 256, 0, stream>>>(T1, DD, e_wk + wdd, DD, e_bk + l * DD, nullptr, Kbuf, DD, DD, 0);
        gemm_wmma_kernel<<<g512, 256, 0, stream>>>(T1, DD, e_wv + wdd, DD, e_bv + l * DD, nullptr, Vbuf, DD, DD, 0);
        attention_kernel<<<gAtt, 128, 0, stream>>>(Q, Kbuf, Vbuf, AO, src, 0);
        gemm_wmma_kernel<<<g512, 256, 0, stream>>>(AO, DD, e_wo + wdd, DD, e_bo + l * DD, X, X, DD, DD, 0);

        layernorm_kernel<<<NT, 256, 0, stream>>>(X, e_ln2_g + l * DD, e_ln2_b + l * DD, T1);
        gemm_wmma_kernel<<<gF,   256, 0, stream>>>(T1,  DD, e_w1 + (size_t)l * DD * FF, FF, e_b1 + l * FF, nullptr, HID, FF, DD, 1);
        gemm_wmma_kernel<<<g512, 256, 0, stream>>>(HID, FF, e_w2 + (size_t)l * FF * DD, DD, e_b2 + l * DD, X, X, DD, FF, 0);
    }
    layernorm_kernel<<<NT, 256, 0, stream>>>(X, e_lnf_g, e_lnf_b, EOUT);

    // ---------------- decoder ----------------
    for (int l = 0; l < LL; ++l) {
        const size_t wdd = (size_t)l * DD * DD;
        // masked self-attention
        layernorm_kernel<<<NT, 256, 0, stream>>>(Y, d_ln1_g + l * DD, d_ln1_b + l * DD, T1);
        gemm_wmma_kernel<<<g512, 256, 0, stream>>>(T1, DD, dsa_wq + wdd, DD, dsa_bq + l * DD, nullptr, Q,    DD, DD, 0);
        gemm_wmma_kernel<<<g512, 256, 0, stream>>>(T1, DD, dsa_wk + wdd, DD, dsa_bk + l * DD, nullptr, Kbuf, DD, DD, 0);
        gemm_wmma_kernel<<<g512, 256, 0, stream>>>(T1, DD, dsa_wv + wdd, DD, dsa_bv + l * DD, nullptr, Vbuf, DD, DD, 0);
        attention_kernel<<<gAtt, 128, 0, stream>>>(Q, Kbuf, Vbuf, AO, trg, 1);
        gemm_wmma_kernel<<<g512, 256, 0, stream>>>(AO, DD, dsa_wo + wdd, DD, dsa_bo + l * DD, Y, Y, DD, DD, 0);

        // cross-attention (K/V from encoder output)
        layernorm_kernel<<<NT, 256, 0, stream>>>(Y, d_ln2_g + l * DD, d_ln2_b + l * DD, T1);
        gemm_wmma_kernel<<<g512, 256, 0, stream>>>(T1,   DD, dca_wq + wdd, DD, dca_bq + l * DD, nullptr, Q,    DD, DD, 0);
        gemm_wmma_kernel<<<g512, 256, 0, stream>>>(EOUT, DD, dca_wk + wdd, DD, dca_bk + l * DD, nullptr, Kbuf, DD, DD, 0);
        gemm_wmma_kernel<<<g512, 256, 0, stream>>>(EOUT, DD, dca_wv + wdd, DD, dca_bv + l * DD, nullptr, Vbuf, DD, DD, 0);
        attention_kernel<<<gAtt, 128, 0, stream>>>(Q, Kbuf, Vbuf, AO, src, 0);
        gemm_wmma_kernel<<<g512, 256, 0, stream>>>(AO, DD, dca_wo + wdd, DD, dca_bo + l * DD, Y, Y, DD, DD, 0);

        // FFN
        layernorm_kernel<<<NT, 256, 0, stream>>>(Y, d_ln3_g + l * DD, d_ln3_b + l * DD, T1);
        gemm_wmma_kernel<<<gF,   256, 0, stream>>>(T1,  DD, d_w1 + (size_t)l * DD * FF, FF, d_b1 + l * FF, nullptr, HID, FF, DD, 1);
        gemm_wmma_kernel<<<g512, 256, 0, stream>>>(HID, FF, d_w2 + (size_t)l * FF * DD, DD, d_b2 + l * DD, Y, Y, DD, FF, 0);
    }
    layernorm_kernel<<<NT, 256, 0, stream>>>(Y, d_lnf_g, d_lnf_b, T1);

    // ---------------- logits + log_softmax ----------------
    gemm_wmma_kernel<<<gV, 256, 0, stream>>>(T1, DD, out_w, VV, out_b, nullptr, OUT, VV, DD, 0);
    logsoftmax_kernel<<<NT, 256, 0, stream>>>(OUT);
}
